// KANC_HDC_88527865905792
// MI455X (gfx1250) — matrix operand order repaired and checked by reading, not compiled, not run on hardware
//
#include <hip/hip_runtime.h>

// ---------------------------------------------------------------------------
// KAN-CNN for MI455X (gfx1250, wave32).
//   - Each unique pixel is spline-expanded ONCE (silu + 8 bases) into a
//     feature-major f16 buffer  ->  kills the 9-12x redundant transcendental
//     VALU work of naive im2col expansion.
//   - Every KANLinear layer is then a pure f16 GEMM on v_wmma_f32_16x16x32_f16
//     with f32 accumulation; A tiles staged in LDS pre-swizzled to a contiguous-k (k=16h+e)
//     16-bit A-fragment layout, weights pre-expanded/pre-swizzled to the
//     B-fragment layout once in workspace.
//   - All layer shapes are template constants: divides become mul-shift,
//     K-loops fully unroll (3 / 36 / 13 WMMAs straight-line per tile).
// ---------------------------------------------------------------------------

typedef __attribute__((ext_vector_type(16))) _Float16 v16h;
typedef __attribute__((ext_vector_type(8)))  float    v8f;

#define NBASES 8   // grid_size(5) + spline_order(3)
#define NFEAT  9   // silu + 8 spline bases

// B-spline bases on the fixed uniform grid g[j] = -1 + (j-3)*0.4, j=0..11.
__device__ __forceinline__ void kan_feats(float x, float f[NFEAT]) {
    f[0] = x / (1.0f + __expf(-x));          // silu
    const float h = 0.4f;
    float b[11];
#pragma unroll
    for (int j = 0; j < 11; ++j) {
        float g0 = -1.0f + (float)(j - 3) * h;
        b[j] = (x >= g0 && x < g0 + h) ? 1.0f : 0.0f;
    }
#pragma unroll
    for (int p = 1; p <= 3; ++p) {
        float inv = 1.0f / ((float)p * h);   // both denominators are p*h (uniform grid)
#pragma unroll
        for (int j = 0; j + p < 11; ++j) {
            float gj   = -1.0f + (float)(j - 3) * h;
            float gjp1 = gj + (float)(p + 1) * h;       // g[j+p+1]
            b[j] = (x - gj) * inv * b[j] + (gjp1 - x) * inv * b[j + 1];
        }
    }
#pragma unroll
    for (int j = 0; j < NBASES; ++j) f[1 + j] = b[j];
}

// Swizzled fragment index (16x32 A tile, contiguous k order).
// lane r   holds M=r, K in {0..15}(e0-15)
// lane r+16 holds M=r, K in {16..31}(e0-15)
__device__ __forceinline__ int a_frag_idx(int r, int kk /*K%32*/) {
    int lane = r + ((kk >> 4) & 1) * 16;
    int el   = kk & 15;
    return (lane << 4) + el;                 // within one 512-half chunk
}

// ---------------------------------------------------------------------------
// Pixel expansion: src (npix f32) -> dst feature-major (NFEAT, npix) f16.
// Coalesced stores per feature plane; each pixel expanded exactly once.
// ---------------------------------------------------------------------------
__global__ void expand_feats(const float* __restrict__ src,
                             _Float16* __restrict__ dst, int npix) {
    int gid = blockIdx.x * blockDim.x + threadIdx.x;
    if (gid >= npix) return;
    float f[NFEAT];
    kan_feats(src[gid], f);
#pragma unroll
    for (int t = 0; t < NFEAT; ++t)
        dst[t * npix + gid] = (_Float16)f[t];
}

// ---------------------------------------------------------------------------
// Weight prep: expand (base_w, spline_w, scaler) -> f16 B-fragments.
// dst layout: [Kpad/32 chunks][32 lanes][16 halves]; lane holds N=lane&15,
// K = chunk*32 + (lane>=16 ? 16 : 0) + e  (contiguous k order, same as the A tile).
// ---------------------------------------------------------------------------
__global__ void prep_kan_w(const float* __restrict__ base_w,
                           const float* __restrict__ spline_w,
                           const float* __restrict__ scaler,
                           _Float16* __restrict__ dst,
                           int in_feats, int out_ch, int nhalves) {
    int gid = blockIdx.x * blockDim.x + threadIdx.x;
    if (gid >= nhalves) return;
    int chunk = gid >> 9;
    int lane  = (gid >> 4) & 31;
    int e     = gid & 15;
    int K = chunk * 32 + ((lane >= 16) ? 16 : 0) + e;
    int n = lane & 15;
    float v = 0.0f;
    if (n < out_ch && K < in_feats * NFEAT) {
        int i = K / NFEAT;
        int f = K - i * NFEAT;
        if (f == 0) v = base_w[n * in_feats + i];
        else        v = spline_w[(n * in_feats + i) * NBASES + (f - 1)] *
                        scaler[n * in_feats + i];
    }
    dst[gid] = (_Float16)v;
}

// FC weights: w1 (200,98) -> [13 ntiles][4 chunks][32 lanes][16 halves]
__global__ void prep_fc_w(const float* __restrict__ w1, _Float16* __restrict__ dst,
                          int nhalves) {
    int gid = blockIdx.x * blockDim.x + threadIdx.x;
    if (gid >= nhalves) return;
    int ntile = gid >> 11;
    int rest  = gid & 2047;
    int chunk = rest >> 9;
    int lane  = (rest >> 4) & 31;
    int e     = rest & 15;
    int K = chunk * 32 + ((lane >= 16) ? 16 : 0) + e;
    int n = ntile * 16 + (lane & 15);
    float v = (K < 98 && n < 200) ? w1[n * 98 + K] : 0.0f;
    dst[gid] = (_Float16)v;
}

// ---------------------------------------------------------------------------
// KAN conv layer: one wave per 16-row tile of the (virtual) im2col matrix.
// ex  : feature-major expanded activations (NFEAT, B*C*H*W) f16
// dst : NCHW f32 output (B, OUTC, Ho, Wo)
// wswz: pre-swizzled f16 B-fragments [KPAD/32][32][16]
// ---------------------------------------------------------------------------
template <int B, int C, int H, int W, int KSZ, int OUTC, int KPAD>
__global__ void kan_conv_wmma(const _Float16* __restrict__ ex,
                              float* __restrict__ dst,
                              const _Float16* __restrict__ wswz) {
    constexpr int Ho   = H - KSZ + 1;
    constexpr int Wo   = W - KSZ + 1;
    constexpr int HoWo = Ho * Wo;
    constexpr int IN   = C * KSZ * KSZ;
    constexpr int NROWS = B * HoWo;
    constexpr int NPIX  = B * C * H * W;
    constexpr int NCHUNK = KPAD / 32;

    __shared__ __align__(32) _Float16 smem[16 * KPAD];
    const int lane = threadIdx.x;                 // 32 threads = 1 wave
    const int row0 = blockIdx.x * 16;

    // warm WGP$/L2 with the (tiny, reused-by-every-block) weight fragments
    __builtin_prefetch(wswz + (lane << 6), 0, 1);

    // zero A tile (K padding + tail rows must contribute 0)
    int* s32 = (int*)smem;
#pragma unroll 4
    for (int i = lane; i < (16 * KPAD) >> 1; i += 32) s32[i] = 0;
    __syncthreads();

    // gather pre-expanded features, scatter into swizzled A tile
    for (int idx = lane; idx < 16 * IN; idx += 32) {
        int r = idx / IN;                          // compile-time divisors ->
        int i = idx - r * IN;                      // mul-shift
        int row = row0 + r;
        if (row >= NROWS) continue;
        int bi  = row / HoWo;
        int rem = row - bi * HoWo;
        int oy  = rem / Wo;
        int ox  = rem - oy * Wo;
        int ci  = i / (KSZ * KSZ);
        int rr  = i - ci * (KSZ * KSZ);
        int kh  = rr / KSZ;
        int kw  = rr - kh * KSZ;
        int p   = ((bi * C + ci) * H + (oy + kh)) * W + (ox + kw);
        int Kb  = i * NFEAT;
#pragma unroll
        for (int t = 0; t < NFEAT; ++t) {
            int K = Kb + t;
            smem[((K >> 5) << 9) + a_frag_idx(r, K & 31)] = ex[t * NPIX + p];
        }
    }
    __syncthreads();

    // fully unrolled K loop: contiguous 32B fragment loads + WMMA accumulate
    v8f acc = {};
#pragma unroll
    for (int c = 0; c < NCHUNK; ++c) {
        v16h a = *(const v16h*)(smem + (c << 9) + (lane << 4));
        v16h b = *(const v16h*)(wswz + (c << 9) + (lane << 4));
        acc = __builtin_amdgcn_wmma_f32_16x16x32_f16(false, a, false, b,
                                                     (short)0, acc, false, false);
    }

    // D layout: lane -> N = lane&15, VGPR v -> M = v + (lane>=16)*8
    int n = lane & 15;
    int hsel = lane >> 4;
    if (n < OUTC) {
#pragma unroll
        for (int v = 0; v < 8; ++v) {
            int row = row0 + v + hsel * 8;
            if (row < NROWS) {
                int bi  = row / HoWo;
                int rem = row - bi * HoWo;
                int oy  = rem / Wo;
                int ox  = rem - oy * Wo;
                dst[((bi * OUTC + n) * Ho + oy) * Wo + ox] = acc[v];
            }
        }
    }
}

// 2x2/2 maxpool over NCHW
__global__ void maxpool2x2(const float* __restrict__ src, float* __restrict__ dst,
                           int total, int Hs, int Ws) {
    int gid = blockIdx.x * blockDim.x + threadIdx.x;
    if (gid >= total) return;
    int Wd = Ws >> 1, Hd = Hs >> 1;
    int px = gid % Wd;
    int t  = gid / Wd;
    int py = t % Hd;
    int bc = t / Hd;
    const float* p = src + ((size_t)bc * Hs + 2 * py) * Ws + 2 * px;
    dst[gid] = fmaxf(fmaxf(p[0], p[1]), fmaxf(p[Ws], p[Ws + 1]));
}

// Dense head: relu(A @ w1^T + b1), A = (1024, 98), out = (1024, 200)
__global__ void fc_wmma(const float* __restrict__ A,
                        const _Float16* __restrict__ wswz,
                        const float* __restrict__ bias,
                        float* __restrict__ out) {
    constexpr int KIN = 98, NOUT = 200, NROWS = 1024;
    __shared__ __align__(32) _Float16 smem[16 * 128];
    const int lane = threadIdx.x;
    const int row0 = blockIdx.x * 16;
    const int ntile = blockIdx.y;
#pragma unroll
    for (int s = 0; s < 64; ++s) {                 // 16*128 / 32
        int idx = lane + s * 32;
        int r = idx >> 7;
        int K = idx & 127;
        int row = row0 + r;
        float v = (K < KIN && row < NROWS) ? A[row * KIN + K] : 0.0f;
        smem[((K >> 5) << 9) + a_frag_idx(r, K & 31)] = (_Float16)v;
    }
    __syncthreads();
    v8f acc = {};
    const _Float16* wp = wswz + (size_t)ntile * 4 * 512;
#pragma unroll
    for (int c = 0; c < 4; ++c) {
        v16h a = *(const v16h*)(smem + (c << 9) + (lane << 4));
        v16h b = *(const v16h*)(wp + (c << 9) + (lane << 4));
        acc = __builtin_amdgcn_wmma_f32_16x16x32_f16(false, a, false, b,
                                                     (short)0, acc, false, false);
    }
    int n = ntile * 16 + (lane & 15);
    int hsel = lane >> 4;
    if (n < NOUT) {
        float bv = bias[n];
#pragma unroll
        for (int vv = 0; vv < 8; ++vv) {
            int row = row0 + vv + hsel * 8;
            if (row < NROWS) {
                float o = acc[vv] + bv;
                out[row * NOUT + n] = o > 0.0f ? o : 0.0f;
            }
        }
    }
}

// ---------------------------------------------------------------------------
// Workspace layout (bytes, 256-aligned, total ~35 MB).  The three expansion
// buffers alias one region (live at disjoint times).
// ---------------------------------------------------------------------------
#define OFF_CONV1  0u                         // (1024,5,26,26) f32 = 13,844,480
#define OFF_H1     13844480u                  // (1024,5,13,13) f32 =  3,461,120
#define OFF_H2     17305600u                  // (1024,5,9,9)   f32 =  1,658,880
#define OFF_H3     18964480u                  // (1024,2,7,7)   f32 =    401,408
#define OFF_EX     19365888u                  // max 9*865280 halves = 15,575,040
#define OFF_W1SWZ  34940928u                  // 3*512 halves   =        3,072
#define OFF_W2SWZ  34944000u                  // 36*512 halves  =       36,864
#define OFF_W3SWZ  34980864u                  // 13*512 halves  =       13,312
#define OFF_FCSWZ  34994176u                  // 13*4*512 halves=       53,248

extern "C" void kernel_launch(void* const* d_in, const int* in_sizes, int n_in,
                              void* d_out, int out_size, void* d_ws, size_t ws_size,
                              hipStream_t stream) {
    const float* x         = (const float*)d_in[0];
    const float* c1_base   = (const float*)d_in[1];
    const float* c1_spline = (const float*)d_in[2];
    const float* c1_scaler = (const float*)d_in[3];
    const float* c2_base   = (const float*)d_in[4];
    const float* c2_spline = (const float*)d_in[5];
    const float* c2_scaler = (const float*)d_in[6];
    const float* c3_base   = (const float*)d_in[7];
    const float* c3_spline = (const float*)d_in[8];
    const float* c3_scaler = (const float*)d_in[9];
    const float* w1        = (const float*)d_in[10];
    const float* b1        = (const float*)d_in[11];

    char* ws = (char*)d_ws;
    float*    conv1f = (float*)(ws + OFF_CONV1);
    float*    h1     = (float*)(ws + OFF_H1);
    float*    h2     = (float*)(ws + OFF_H2);
    float*    h3     = (float*)(ws + OFF_H3);
    _Float16* ex     = (_Float16*)(ws + OFF_EX);
    _Float16* w1swz  = (_Float16*)(ws + OFF_W1SWZ);
    _Float16* w2swz  = (_Float16*)(ws + OFF_W2SWZ);
    _Float16* w3swz  = (_Float16*)(ws + OFF_W3SWZ);
    _Float16* fcswz  = (_Float16*)(ws + OFF_FCSWZ);

    // --- weight expansion + fragment swizzle (tiny; stays hot in L2/WGP$) ---
    prep_kan_w<<<(1536  + 255) / 256, 256, 0, stream>>>(c1_base, c1_spline, c1_scaler,
                                                        w1swz, 9,   5, 1536);
    prep_kan_w<<<(18432 + 255) / 256, 256, 0, stream>>>(c2_base, c2_spline, c2_scaler,
                                                        w2swz, 125, 5, 18432);
    prep_kan_w<<<(6656  + 255) / 256, 256, 0, stream>>>(c3_base, c3_spline, c3_scaler,
                                                        w3swz, 45,  2, 6656);
    prep_fc_w <<<(26624 + 255) / 256, 256, 0, stream>>>(w1, fcswz, 26624);

    // --- layer 1: expand x once, then WMMA conv (1,28,28)->(5,26,26) ---
    {
        int npix = 1024 * 28 * 28;                        // 802816
        expand_feats<<<(npix + 255) / 256, 256, 0, stream>>>(x, ex, npix);
        int nrows = 1024 * 26 * 26;                       // 692224
        kan_conv_wmma<1024, 1, 28, 28, 3, 5, 96>
            <<<(nrows + 15) / 16, 32, 0, stream>>>(ex, conv1f, w1swz);
    }
    // --- maxpool 2x2 -> (1024,5,13,13) ---
    {
        int total = 1024 * 5 * 13 * 13;
        maxpool2x2<<<(total + 255) / 256, 256, 0, stream>>>(conv1f, h1, total, 26, 26);
    }
    // --- layer 2: (5,13,13)->(5,9,9), K=1125->1152 ---
    {
        int npix = 1024 * 5 * 13 * 13;                    // 865280
        expand_feats<<<(npix + 255) / 256, 256, 0, stream>>>(h1, ex, npix);
        int nrows = 1024 * 9 * 9;                         // 82944
        kan_conv_wmma<1024, 5, 13, 13, 5, 5, 1152>
            <<<(nrows + 15) / 16, 32, 0, stream>>>(ex, h2, w2swz);
    }
    // --- layer 3: (5,9,9)->(2,7,7), K=405->416 ---
    {
        int npix = 1024 * 5 * 9 * 9;                      // 414720
        expand_feats<<<(npix + 255) / 256, 256, 0, stream>>>(h2, ex, npix);
        int nrows = 1024 * 7 * 7;                         // 50176
        kan_conv_wmma<1024, 5, 9, 9, 3, 2, 416>
            <<<(nrows + 15) / 16, 32, 0, stream>>>(ex, h3, w3swz);
    }
    // --- FC head: relu((1024,98) @ (98,200)^T + b1) -> d_out ---
    fc_wmma<<<dim3(64, 13), 32, 0, stream>>>(h3, fcswz, b1, (float*)d_out);
}
